// SimCSEModel_53386443489488
// MI455X (gfx1250) — matrix operand order repaired and checked by reading, not compiled, not run on hardware
//
#include <hip/hip_runtime.h>
#include <hip/hip_bf16.h>

typedef __bf16 v16bf __attribute__((ext_vector_type(16)));
typedef __bf16 v8bf  __attribute__((ext_vector_type(8)));
typedef float  v8f   __attribute__((ext_vector_type(8)));

#define BN 8192
#define DN 256
#define INV_T 20.0f   // 1 / 0.05

// ---------------------------------------------------------------------------
// Kernel 1: row L2-normalization + bf16 hi/lo split.  One wave32 per row.
// x = hi + lo with hi = bf16(x), lo = bf16(x - hi)  ->  ~16 mantissa bits.
// ---------------------------------------------------------------------------
__global__ __launch_bounds__(256) void normalize_split(const float* __restrict__ in1,
                                                       const float* __restrict__ in2,
                                                       __bf16* __restrict__ h1,
                                                       __bf16* __restrict__ l1,
                                                       __bf16* __restrict__ h2,
                                                       __bf16* __restrict__ l2) {
    const int wavesPerBlock = blockDim.x >> 5;
    const int wid  = blockIdx.x * wavesPerBlock + (threadIdx.x >> 5);
    const int lane = threadIdx.x & 31;

    const float* src;
    __bf16 *dh, *dl;
    int row;
    if (wid < BN) { src = in1; dh = h1; dl = l1; row = wid; }
    else          { src = in2; dh = h2; dl = l2; row = wid - BN; }

    const float* p = src + (size_t)row * DN;
    float v[8];
    float ss = 0.0f;
    #pragma unroll
    for (int k = 0; k < 8; ++k) {
        v[k] = p[lane + 32 * k];
        ss += v[k] * v[k];
    }
    #pragma unroll
    for (int off = 16; off >= 1; off >>= 1)
        ss += __shfl_xor(ss, off, 32);

    const float inv = 1.0f / fmaxf(sqrtf(ss), 1e-8f);
    __bf16* qh = dh + (size_t)row * DN;
    __bf16* ql = dl + (size_t)row * DN;
    #pragma unroll
    for (int k = 0; k < 8; ++k) {
        const float xn = v[k] * inv;
        const __bf16 hi = (__bf16)xn;
        const __bf16 lo = (__bf16)(xn - (float)hi);
        qh[lane + 32 * k] = hi;
        ql[lane + 32 * k] = lo;
    }
}

// ---------------------------------------------------------------------------
// Kernel 2: split-bf16 WMMA GEMM + single-pass fused log-softmax loss.
// Block = 256 threads (8 waves), owns 16 rows. Waves stripe the 512 col-tiles.
// sim = Ah*Bh + Ah*Bl + Al*Bh  (error ~2^-16, well inside fp32 tolerance)
// ---------------------------------------------------------------------------
__global__ __launch_bounds__(256) void simcse_loss(const __bf16* __restrict__ n1h,
                                                   const __bf16* __restrict__ n1l,
                                                   const __bf16* __restrict__ n2h,
                                                   const __bf16* __restrict__ n2l,
                                                   const float*  __restrict__ weights,
                                                   const int*    __restrict__ mask,
                                                   float* __restrict__ out) {
    __shared__ __attribute__((aligned(16))) __bf16 sAh[16 * DN];  // 8 KB
    __shared__ __attribute__((aligned(16))) __bf16 sAl[16 * DN];  // 8 KB
    __shared__ float rS[16], rT[16], rW[16], rC[16];

    const int tid = threadIdx.x;
    const int i0  = blockIdx.x * 16;

    // Stage A-tile hi/lo into LDS (reused for all 512 column tiles).
    for (int idx = tid; idx < 16 * DN; idx += 256) {
        const int r = idx >> 8;          // idx / 256
        const int c = idx & (DN - 1);
        const size_t g = (size_t)(i0 + r) * DN + c;
        sAh[idx] = n1h[g];
        sAl[idx] = n1l[g];
    }
    if (tid < 16) { rS[tid] = 0.0f; rT[tid] = 0.0f; rW[tid] = 0.0f; rC[tid] = 0.0f; }
    __syncthreads();

    const int lane    = tid & 31;
    const int wave    = tid >> 5;
    const int laneLow = lane & 15;   // A row / B col index within tile
    const int laneHi  = lane >> 4;   // selects K half (A/B) or M half (C)

    float Sacc[8] = {0}, Tacc[8] = {0}, Wacc[8] = {0}, Cacc[8] = {0};

    // 16-bit A 16x32 layout: lane holds K = 8*laneHi+[0..7] and 16+8*laneHi+[0..7].
    const __bf16* aHb = &sAh[laneLow * DN + 8 * laneHi];
    const __bf16* aLb = &sAl[laneLow * DN + 8 * laneHi];

    for (int jt = wave; jt < BN / 16; jt += 8) {
        const int j0 = jt * 16;
        // 16-bit B 32x16 layout: lane = col N (n2 row j0+N), K = 8*laneHi+[0..7] and 16+8*laneHi+[0..7].
        const size_t brow = (size_t)(j0 + laneLow) * DN + 8 * laneHi;
        const __bf16* bHb = n2h + brow;
        const __bf16* bLb = n2l + brow;

        v8f c = {0.f, 0.f, 0.f, 0.f, 0.f, 0.f, 0.f, 0.f};
        #pragma unroll
        for (int kc = 0; kc < DN / 32; ++kc) {          // 8 K-chunks of 32
            v16bf ah, al, bh, bl;
            *(v8bf*)&ah       = *(const v8bf*)(aHb + 32 * kc);
            *((v8bf*)&ah + 1) = *(const v8bf*)(aHb + 32 * kc + 16);
            *(v8bf*)&al       = *(const v8bf*)(aLb + 32 * kc);
            *((v8bf*)&al + 1) = *(const v8bf*)(aLb + 32 * kc + 16);
            *(v8bf*)&bh       = *(const v8bf*)(bHb + 32 * kc);
            *((v8bf*)&bh + 1) = *(const v8bf*)(bHb + 32 * kc + 16);
            *(v8bf*)&bl       = *(const v8bf*)(bLb + 32 * kc);
            *((v8bf*)&bl + 1) = *(const v8bf*)(bLb + 32 * kc + 16);
            c = __builtin_amdgcn_wmma_f32_16x16x32_bf16(false, ah, false, bh,
                                                        (short)0, c, false, false);
            c = __builtin_amdgcn_wmma_f32_16x16x32_bf16(false, ah, false, bl,
                                                        (short)0, c, false, false);
            c = __builtin_amdgcn_wmma_f32_16x16x32_bf16(false, al, false, bh,
                                                        (short)0, c, false, false);
        }

        // C layout: VGPR r, lanes 0-15 -> M=r; lanes 16-31 -> M=8+r; N=lane%16.
        const int col = j0 + laneLow;
        #pragma unroll
        for (int r = 0; r < 8; ++r) {
            const int row = i0 + r + 8 * laneHi;
            const float s = c[r] * INV_T;               // sim value, in [-20, 20]
            Sacc[r] += __expf(s);
            const size_t off = (size_t)row * BN + col;
            const float w = __builtin_nontemporal_load(&weights[off]);
            const int   m = __builtin_nontemporal_load(&mask[off]);
            const float mw = m ? w : 0.0f;
            Tacc[r] += mw * s;
            Wacc[r] += mw;
            Cacc[r] += (float)m;
        }
    }

    // Reduce across the 16 column-lanes of each half-wave, then across waves.
    #pragma unroll
    for (int r = 0; r < 8; ++r) {
        float S = Sacc[r], T = Tacc[r], W = Wacc[r], C = Cacc[r];
        #pragma unroll
        for (int off = 8; off >= 1; off >>= 1) {
            S += __shfl_xor(S, off, 32);
            T += __shfl_xor(T, off, 32);
            W += __shfl_xor(W, off, 32);
            C += __shfl_xor(C, off, 32);
        }
        if (laneLow == 0) {                  // lanes 0 and 16 hold the row sums
            const int rl = r + 8 * laneHi;
            atomicAdd(&rS[rl], S);
            atomicAdd(&rT[rl], T);
            atomicAdd(&rW[rl], W);
            atomicAdd(&rC[rl], C);
        }
    }
    __syncthreads();

    // Per-row loss, then block sum -> one global atomic per block.
    if (tid < 32) {
        float li = 0.0f;
        if (tid < 16) {
            const float lse = logf(rS[tid]);
            const float cnt = fmaxf(rC[tid], 1.0f);
            li = -(rT[tid] - rW[tid] * lse) / cnt;
        }
        #pragma unroll
        for (int off = 8; off >= 1; off >>= 1)
            li += __shfl_xor(li, off, 32);
        if (tid == 0) atomicAdd(out, li);
    }
}

// ---------------------------------------------------------------------------
extern "C" void kernel_launch(void* const* d_in, const int* in_sizes, int n_in,
                              void* d_out, int out_size, void* d_ws, size_t ws_size,
                              hipStream_t stream) {
    const float* e1 = (const float*)d_in[0];   // encoded_texts_1 [8192,256] f32
    const float* e2 = (const float*)d_in[1];   // encoded_texts_2 [8192,256] f32
    const float* w  = (const float*)d_in[2];   // weights [8192,8192] f32
    const int*   m  = (const int*)d_in[3];     // mask    [8192,8192] i32
    float* out = (float*)d_out;                // scalar f32

    // Workspace: 4 x (8192*256) bf16 = 16 MB
    __bf16* h1 = (__bf16*)d_ws;
    __bf16* l1 = h1 + (size_t)BN * DN;
    __bf16* h2 = l1 + (size_t)BN * DN;
    __bf16* l2 = h2 + (size_t)BN * DN;

    hipMemsetAsync(out, 0, sizeof(float), stream);
    normalize_split<<<(2 * BN) / 8, 256, 0, stream>>>(e1, e2, h1, l1, h2, l2);
    simcse_loss<<<BN / 16, 256, 0, stream>>>(h1, l1, h2, l2, w, m, out);
}
